// _HessianAttn_72868415144567
// MI455X (gfx1250) — compile-verified, matrix operand order checked by reading, not run on hardware
//
#include <hip/hip_runtime.h>
#include <math.h>

// ---------------- CDNA5 WMMA types / helpers ----------------
typedef __attribute__((ext_vector_type(16))) __bf16 v16bf;
typedef __attribute__((ext_vector_type(8)))  __bf16 v8bf;
typedef __attribute__((ext_vector_type(4)))  __bf16 v4bf;
typedef __attribute__((ext_vector_type(8)))  float  v8f;

__device__ __forceinline__ v8f wmma_bf16(v16bf a, v16bf b, v8f c) {
  // D = A(16x32 bf16) * B(32x16 bf16) + C(16x16 f32)
  return __builtin_amdgcn_wmma_f32_16x16x32_bf16(false, a, false, b,
                                                 (short)0, c, false, false);
}

__device__ __forceinline__ v16bf frag_from_chunks(v8bf c0, v8bf c1) {
  v16bf r;
#pragma unroll
  for (int i = 0; i < 8; ++i) { r[i] = c0[i]; r[i + 8] = c1[i]; }
  return r;
}

// 16-bit operand fragment: lane holds row/col (lane%16); element e maps to
// k = (e%8) + (e/8)*16 + (lane/16)*8  ->  two contiguous 8-elem (16B) chunks.
__device__ __forceinline__ v16bf load_frag_bf16(const __bf16* p, int k0, int half) {
  v8bf c0 = *reinterpret_cast<const v8bf*>(p + k0 + half * 8);
  v8bf c1 = *reinterpret_cast<const v8bf*>(p + k0 + 16 + half * 8);
  return frag_from_chunks(c0, c1);
}

// ---------------- elementwise f32 -> bf16 (vec4) ----------------
__global__ __launch_bounds__(256) void cvt_bf16(const float* __restrict__ src,
                                                __bf16* __restrict__ dst) {
  const size_t i = ((size_t)blockIdx.x * 256 + threadIdx.x) * 4;
  float4 f = *(const float4*)(src + i);
  v4bf o;
  o[0] = (__bf16)f.x; o[1] = (__bf16)f.y; o[2] = (__bf16)f.z; o[3] = (__bf16)f.w;
  *(v4bf*)(dst + i) = o;
}

// ---------------- GEMM: Y[MxN] = X[MxK] @ W[NxK]^T (bf16 in, f32 out) ----------------
// 128 threads = 4 waves stacked in M; wave computes 32x64 (2x4 WMMA subtiles):
// 8 WMMAs per 12 b128 loads per k-step of 32.
__global__ __launch_bounds__(128) void gemm_bf16_nt(const __bf16* __restrict__ X,
                                                    const __bf16* __restrict__ W,
                                                    float* __restrict__ Y,
                                                    int M, int N, int K) {
  const int wave = threadIdx.x >> 5;
  const int lane = threadIdx.x & 31;
  const int half = lane >> 4;
  const int lm   = lane & 15;
  const int row0 = blockIdx.y * 128 + wave * 32;
  const int col0 = blockIdx.x * 64;

  const __bf16* x0 = X + (size_t)(row0 + lm) * K;
  const __bf16* x1 = X + (size_t)(row0 + 16 + lm) * K;
  const __bf16* w0 = W + (size_t)(col0 + lm) * K;
  const __bf16* w1 = w0 + (size_t)16 * K;
  const __bf16* w2 = w0 + (size_t)32 * K;
  const __bf16* w3 = w0 + (size_t)48 * K;

  v8f acc[2][4] = {};
  for (int k0 = 0; k0 < K; k0 += 32) {
    __builtin_prefetch(x0 + k0 + 256, 0, 3);   // global_prefetch_b8, ~8 iters ahead
    __builtin_prefetch(w0 + k0 + 256, 0, 3);
    v16bf a0 = load_frag_bf16(x0, k0, half);
    v16bf a1 = load_frag_bf16(x1, k0, half);
    v16bf b0 = load_frag_bf16(w0, k0, half);
    v16bf b1 = load_frag_bf16(w1, k0, half);
    v16bf b2 = load_frag_bf16(w2, k0, half);
    v16bf b3 = load_frag_bf16(w3, k0, half);
    acc[0][0] = wmma_bf16(a0, b0, acc[0][0]);
    acc[0][1] = wmma_bf16(a0, b1, acc[0][1]);
    acc[0][2] = wmma_bf16(a0, b2, acc[0][2]);
    acc[0][3] = wmma_bf16(a0, b3, acc[0][3]);
    acc[1][0] = wmma_bf16(a1, b0, acc[1][0]);
    acc[1][1] = wmma_bf16(a1, b1, acc[1][1]);
    acc[1][2] = wmma_bf16(a1, b2, acc[1][2]);
    acc[1][3] = wmma_bf16(a1, b3, acc[1][3]);
  }
#pragma unroll
  for (int ti = 0; ti < 2; ++ti) {
#pragma unroll
    for (int r = 0; r < 8; ++r) {
      const size_t row = (size_t)(row0 + ti * 16 + r + 8 * half) * N + col0 + lm;
#pragma unroll
      for (int tj = 0; tj < 4; ++tj) Y[row + tj * 16] = acc[ti][tj][r];
    }
  }
}

// ---------------- RMSNorm + RoPE (+gain), f32 -> bf16 ----------------
// one wave per (row m, head h); lane handles pair indices j=2*lane, 2*lane+1.
__global__ __launch_bounds__(128) void norm_rope(const float* __restrict__ src,
                                                 __bf16* __restrict__ dst,
                                                 const float* __restrict__ gain,
                                                 int nh, int S, float base) {
  const int wave = threadIdx.x >> 5;
  const int lane = threadIdx.x & 31;
  const int id = blockIdx.x * 4 + wave;
  const int m = id / nh;
  const int h = id - m * nh;
  const float* p = src + ((size_t)m * nh + h) * 128;

  const int j = lane * 2;
  float x1a = p[j],      x1b = p[j + 1];
  float x2a = p[j + 64], x2b = p[j + 65];
  float ss = x1a * x1a + x1b * x1b + x2a * x2a + x2b * x2b;
#pragma unroll
  for (int d = 16; d >= 1; d >>= 1) ss += __shfl_xor(ss, d, 32);
  float inv = rsqrtf(ss * (1.0f / 128.0f) + 1.1920929e-07f);
  if (gain) inv *= gain[h];  // per-head scalar commutes with rotation

  const int t = m % S;
  const float lb = logf(base);
  const float fa = expf(-((float)j / 64.0f) * lb);
  const float fb = expf(-((float)(j + 1) / 64.0f) * lb);
  float sa, ca, sb, cb;
  sincosf((float)t * fa, &sa, &ca);
  sincosf((float)t * fb, &sb, &cb);

  x1a *= inv; x1b *= inv; x2a *= inv; x2b *= inv;
  __bf16* q = dst + ((size_t)m * nh + h) * 128;
  q[j]      = (__bf16)( x1a * ca + x2a * sa);
  q[j + 1]  = (__bf16)( x1b * cb + x2b * sb);
  q[j + 64] = (__bf16)(-x1a * sa + x2a * ca);
  q[j + 65] = (__bf16)(-x1b * sb + x2b * cb);
}

// ---------------- V transpose: [B*S, KH*128] f32 -> [B][KH][128][S] bf16 ----------------
__global__ __launch_bounds__(256) void vtrans(const float* __restrict__ src,
                                              __bf16* __restrict__ dst, int S) {
  __shared__ float tile[32][33];
  const int bk = blockIdx.z;          // b*KH + kvh
  const int s0 = blockIdx.x * 32;
  const int c0 = blockIdx.y * 32;
  const int b = bk >> 2, kvh = bk & 3;
  const float* in = src + (size_t)(b * S) * 512 + kvh * 128;
#pragma unroll
  for (int i = threadIdx.y; i < 32; i += 8)
    tile[i][threadIdx.x] = in[(size_t)(s0 + i) * 512 + c0 + threadIdx.x];
  __syncthreads();
  __bf16* out = dst + (size_t)bk * 128 * S;
#pragma unroll
  for (int i = threadIdx.y; i < 32; i += 8)
    out[(size_t)(c0 + i) * S + s0 + threadIdx.x] = (__bf16)tile[threadIdx.x][i];
}

// ---------------- Causal flash attention (GQA 16q/4kv, hd=128) ----------------
// grid (S/64, H, B); 4 waves/block; wave owns 16 q-rows, streams 32-key blocks.
// Output written as bf16 [M, H*128] so the Wproj GEMM consumes bf16 directly.
__global__ __launch_bounds__(128) void attn_fwd(const __bf16* __restrict__ qb,
                                                const __bf16* __restrict__ kb,
                                                const __bf16* __restrict__ vT,
                                                __bf16* __restrict__ out) {
  constexpr int S = 2048, H = 16, KH = 4, HD = 128;
  const float scale = 0.08838834764831845f;  // 1/sqrt(128)
  __shared__ __bf16 pstage[4][16][32];

  const int wave = threadIdx.x >> 5;
  const int lane = threadIdx.x & 31;
  const int half = lane >> 4;
  const int lm   = lane & 15;
  const int b = blockIdx.z, h = blockIdx.y, kvh = h >> 2;
  const int q0 = blockIdx.x * 64 + wave * 16;

  const __bf16* qrow = qb + ((size_t)(b * S + q0 + lm) * H + h) * HD;
  v16bf qf[4];
#pragma unroll
  for (int kc = 0; kc < 4; ++kc) qf[kc] = load_frag_bf16(qrow, kc * 32, half);

  v8f o[8] = {};
  float mstat[8], lstat[8];
#pragma unroll
  for (int r = 0; r < 8; ++r) { mstat[r] = -1e30f; lstat[r] = 0.0f; }

  const int kend = q0 + 15;
  for (int key0 = 0; key0 <= kend; key0 += 32) {
    const __bf16* kr0 = kb + ((size_t)(b * S + key0 + lm) * KH + kvh) * HD;
    const __bf16* kr1 = kb + ((size_t)(b * S + key0 + 16 + lm) * KH + kvh) * HD;
    v8f s0 = {}, s1 = {};
#pragma unroll
    for (int kc = 0; kc < 4; ++kc) {
      s0 = wmma_bf16(qf[kc], load_frag_bf16(kr0, kc * 32, half), s0);
      s1 = wmma_bf16(qf[kc], load_frag_bf16(kr1, kc * 32, half), s1);
    }
    const int t0 = key0 + lm, t1 = key0 + 16 + lm;
#pragma unroll
    for (int r = 0; r < 8; ++r) {
      const int mrow = q0 + r + 8 * half;
      float v0 = (t0 <= mrow) ? s0[r] * scale : -1e30f;
      float v1 = (t1 <= mrow) ? s1[r] * scale : -1e30f;
      float mx = fmaxf(v0, v1);
#pragma unroll
      for (int d = 8; d >= 1; d >>= 1) mx = fmaxf(mx, __shfl_xor(mx, d, 32));
      const float mnew  = fmaxf(mstat[r], mx);
      const float alpha = __expf(mstat[r] - mnew);
      const float p0 = __expf(v0 - mnew);
      const float p1 = __expf(v1 - mnew);
      float rs = p0 + p1;
#pragma unroll
      for (int d = 8; d >= 1; d >>= 1) rs += __shfl_xor(rs, d, 32);
      lstat[r] = lstat[r] * alpha + rs;
      mstat[r] = mnew;
#pragma unroll
      for (int n = 0; n < 8; ++n) o[n][r] *= alpha;
      pstage[wave][r + 8 * half][lm]      = (__bf16)p0;
      pstage[wave][r + 8 * half][16 + lm] = (__bf16)p1;
    }
    asm volatile("s_wait_dscnt 0" ::: "memory");  // P tile visible (intra-wave)
    v8bf pc0 = *reinterpret_cast<const v8bf*>(&pstage[wave][lm][half * 8]);
    v8bf pc1 = *reinterpret_cast<const v8bf*>(&pstage[wave][lm][16 + half * 8]);
    v16bf pf = frag_from_chunks(pc0, pc1);
#pragma unroll
    for (int n = 0; n < 8; ++n) {
      const __bf16* vrow = vT + ((size_t)(b * KH + kvh) * HD + n * 16 + lm) * S + key0;
      v8bf c0 = *reinterpret_cast<const v8bf*>(vrow + half * 8);
      v8bf c1 = *reinterpret_cast<const v8bf*>(vrow + 16 + half * 8);
      o[n] = wmma_bf16(pf, frag_from_chunks(c0, c1), o[n]);
    }
  }
#pragma unroll
  for (int r = 0; r < 8; ++r) {
    const float inv = 1.0f / lstat[r];
    const size_t rowoff = (size_t)(b * S + q0 + r + 8 * half) * (H * HD) + h * HD;
#pragma unroll
    for (int n = 0; n < 8; ++n)
      out[rowoff + n * 16 + lm] = (__bf16)(o[n][r] * inv);
  }
}

// ---------------- launcher ----------------
extern "C" void kernel_launch(void* const* d_in, const int* in_sizes, int n_in,
                              void* d_out, int out_size, void* d_ws, size_t ws_size,
                              hipStream_t stream) {
  const float* x     = (const float*)d_in[0];
  const float* Wq    = (const float*)d_in[1];
  const float* Wk    = (const float*)d_in[2];
  const float* Wv    = (const float*)d_in[3];
  const float* Wproj = (const float*)d_in[4];
  const float* qgain = (const float*)d_in[5];

  constexpr int B = 2, S = 2048, D = 2048, H = 16, KH = 4;
  constexpr int M = B * S;           // 4096
  constexpr int KVD = KH * 128;      // 512

  char* ws = (char*)d_ws;
  __bf16* xb    = (__bf16*)(ws);                    // 16 MB  x as bf16
  __bf16* wb    = (__bf16*)(ws + 16777216ull);      //  8 MB  staged weight (reused 4x)
  float*  Qf    = (float*)(ws + 25165824ull);       // 32 MB  f32 Q (region reused by attnb)
  float*  Kf    = (float*)(ws + 58720256ull);       //  8 MB
  float*  Vf    = (float*)(ws + 67108864ull);       //  8 MB
  __bf16* qb    = (__bf16*)(ws + 75497472ull);      // 16 MB
  __bf16* kb    = (__bf16*)(ws + 92274688ull);      //  4 MB
  __bf16* vT    = (__bf16*)(ws + 96468992ull);      //  4 MB   (total ~96 MB)
  __bf16* attnb = (__bf16*)(ws + 25165824ull);      // 16 MB, overlays Qf (dead by then)

  // 0) one-time bf16 conversions (x up front; weights staged per-GEMM)
  cvt_bf16<<<(M * D) / 1024, 256, 0, stream>>>(x, xb);

  // 1) QKV projections (Y = X @ W^T), bf16 WMMA, f32 accumulate/out
  cvt_bf16<<<(D * D) / 1024, 256, 0, stream>>>(Wq, wb);
  gemm_bf16_nt<<<dim3(D / 64, M / 128), 128, 0, stream>>>(xb, wb, Qf, M, D, D);
  cvt_bf16<<<(KVD * D) / 1024, 256, 0, stream>>>(Wk, wb);
  gemm_bf16_nt<<<dim3(KVD / 64, M / 128), 128, 0, stream>>>(xb, wb, Kf, M, KVD, D);
  cvt_bf16<<<(KVD * D) / 1024, 256, 0, stream>>>(Wv, wb);
  gemm_bf16_nt<<<dim3(KVD / 64, M / 128), 128, 0, stream>>>(xb, wb, Vf, M, KVD, D);

  // 2) RMSNorm + RoPE (+gain) -> bf16; V transpose -> bf16
  const float base = 10000.0f * powf(2.0f, 128.0f / 126.0f);  // S=2048 > 1024
  norm_rope<<<(M * H) / 4, 128, 0, stream>>>(Qf, qb, qgain, H, S, base);
  norm_rope<<<(M * KH) / 4, 128, 0, stream>>>(Kf, kb, nullptr, KH, S, base);
  vtrans<<<dim3(S / 32, 128 / 32, B * KH), dim3(32, 8), 0, stream>>>(Vf, vT, S);

  // 3) causal flash attention -> attnb (bf16, [M, D])
  attn_fwd<<<dim3(S / 64, H, B), 128, 0, stream>>>(qb, kb, vT, attnb);

  // 4) output projection -> d_out (f32)
  cvt_bf16<<<(D * D) / 1024, 256, 0, stream>>>(Wproj, wb);
  gemm_bf16_nt<<<dim3(D / 64, M / 128), 128, 0, stream>>>(attnb, wb, (float*)d_out, M, D, D);
}
